// RealNVP_76776835383756
// MI455X (gfx1250) — matrix operand order repaired and checked by reading, not compiled, not run on hardware
//
#include <hip/hip_runtime.h>
#include <hip/hip_bf16.h>

// ---------------------------------------------------------------------------
// RealNVP fused forward for MI455X (gfx1250, wave32, WMMA).
//
// - All 8 coupling layers fused; hidden activations never touch HBM.
// - Each wave owns a 16-row strip end-to-end -> zero barriers.
// - GEMMs on v_wmma_f32_16x16x32_bf16 (f32 accumulate).
// - Weights pre-packed once per call into bf16 B-fragment layout in d_ws
//   (9.4 MB -> L2 resident); ~16 flops per L2 byte, amplified by WGP$ since
//   all co-resident waves stream identical weight addresses in near-lockstep.
// - Single LDS activation buffer (GEMM2 writes in place; its A operand is
//   fully register-cached) -> ~21 KB LDS/block -> high blocks/WGP.
// ---------------------------------------------------------------------------

typedef __attribute__((ext_vector_type(16))) __bf16 v16bf;
typedef __attribute__((ext_vector_type(8)))  float  v8f;

struct U32x8 { uint4 lo, hi; };   // 32 bytes == one 16x bf16 fragment

__device__ __forceinline__ unsigned short f2bf_rne(float f) {  // weights (one-shot)
  unsigned int u = __float_as_uint(f);
  u += 0x7FFFu + ((u >> 16) & 1u);
  return (unsigned short)(u >> 16);
}

__device__ __forceinline__ v16bf bitcast16(uint4 lo, uint4 hi) {
  U32x8 s; s.lo = lo; s.hi = hi;
  return __builtin_bit_cast(v16bf, s);
}

// B fragment: packed layout, lane-contiguous 32B -> 2x global_load_b128
__device__ __forceinline__ v16bf ldB(const unsigned short* fragBase, int lane) {
  const uint4* p = (const uint4*)(fragBase + lane * 16);
  return bitcast16(p[0], p[1]);
}

// A fragment from row-major bf16 LDS (row = lane&15).
// Contiguous k order (matches packed B): half0 = K0..15, half1 = K16..31.
__device__ __forceinline__ v16bf ldA(const unsigned short* rowBase, int kt, int hf) {
  const unsigned short* p = rowBase + kt * 32 + hf * 16;
  return bitcast16(*(const uint4*)p, *(const uint4*)(p + 8));
}

__device__ __forceinline__ v8f wmma_bf16(v16bf a, v16bf b, v8f c) {
  return __builtin_amdgcn_wmma_f32_16x16x32_bf16(false, a, false, b, (short)0, c,
                                                 false, false);
}

// relu in one v_med3_f32, then round-half-up; caller stores high 16 bits
__device__ __forceinline__ unsigned int relu_bf16_bits(float x) {
  float r = __builtin_amdgcn_fmed3f(x, 0.0f, __builtin_inff());
  return __float_as_uint(r) + 0x8000u;
}

// ---------------------------------------------------------------------------
// Weight packer: f32 row-major (L,K,N) -> bf16 WMMA B-fragments.
// Fragment (nt,kt): 32 lanes x 16 bf16; lane = col nt*16+(l&15);
// element j = W[kt*32 + (l>>4)*16 + j][col].
// ---------------------------------------------------------------------------
__global__ void pack_w(const float* __restrict__ src, unsigned short* __restrict__ dst,
                       int K, int N, int dstLayerStride) {
  int tid  = blockIdx.x * blockDim.x + threadIdx.x;
  int lane = tid & 31;
  int frag = tid >> 5;
  int KT = K >> 5, NT = N >> 4;
  int fpl = KT * NT;
  int layer = frag / fpl;
  if (layer >= 8) return;
  int fl = frag - layer * fpl;
  int nt = fl / KT, kt = fl - nt * KT;
  int col   = nt * 16 + (lane & 15);
  int kbase = kt * 32 + (lane >> 4) * 16;
  const float* s = src + (size_t)layer * K * N;
  unsigned int w[8];
#pragma unroll
  for (int j = 0; j < 8; ++j) {
    unsigned int a = f2bf_rne(s[(size_t)(kbase + 2 * j)     * N + col]);
    unsigned int b = f2bf_rne(s[(size_t)(kbase + 2 * j + 1) * N + col]);
    w[j] = a | (b << 16);
  }
  uint4* d = (uint4*)(dst + (size_t)layer * dstLayerStride + (size_t)fl * 512 + lane * 16);
  d[0] = make_uint4(w[0], w[1], w[2], w[3]);
  d[1] = make_uint4(w[4], w[5], w[6], w[7]);
}

// ---------------------------------------------------------------------------
// Fused flow kernel: 1 wave per block, 16 rows per block, ~21 KB LDS.
// ---------------------------------------------------------------------------
#define ROWS 16
#define HSTR 520   // 512 + 8 bf16 pad -> 1040B row stride
#define ZSTR 68    // 64 + 4 f32 pad

__global__ __launch_bounds__(32)
void realnvp_fused(const float* __restrict__ y,
                   const float* __restrict__ bt1, const float* __restrict__ bt2,
                   const float* __restrict__ bt3,
                   const float* __restrict__ bs1, const float* __restrict__ bs2,
                   const float* __restrict__ bs3,
                   const float* __restrict__ s_scale, const float* __restrict__ s_shift,
                   const unsigned short* __restrict__ pw,
                   float* __restrict__ outZ, float* __restrict__ outLD)
{
  __shared__ float          zbuf[ROWS][ZSTR];
  __shared__ unsigned short h[ROWS][HSTR];     // single buffer: h1 then h2 in place

  const int lane = threadIdx.x;       // 0..31
  const int hf   = lane >> 4;
  const int ln   = lane & 15;
  const int row0 = blockIdx.x * ROWS;

  // ---- load 16x64 z tile ----
  {
    int r  = lane >> 1;
    int cb = (lane & 1) * 32;
    const float4* src = (const float4*)(y + (size_t)(row0 + r) * 64 + cb);
    float4* dst = (float4*)&zbuf[r][cb];
#pragma unroll
    for (int j = 0; j < 8; ++j) dst[j] = src[j];
  }

  float logacc[8];
#pragma unroll
  for (int v = 0; v < 8; ++v) logacc[v] = 0.0f;

#pragma unroll 1
  for (int i = 0; i < 8; ++i) {
    const int off0 = (i & 1);       // transformed-half column offset
    const int off1 = off0 ^ 1;      // conditioning-half column offset

    // ---- stage conditioning half z1 (16x32) as bf16 into h[.][0..31] ----
    {
      unsigned int* drow = (unsigned int*)&h[ln][hf * 16];
#pragma unroll
      for (int j = 0; j < 16; j += 2) {
        int c = hf * 16 + j;
        unsigned int a = (__float_as_uint(zbuf[ln][2 * c + off1]) + 0x8000u) >> 16;
        unsigned int b = (__float_as_uint(zbuf[ln][2 * (c + 1) + off1]) + 0x8000u)
                         & 0xFFFF0000u;
        drow[j >> 1] = a | b;
      }
    }
    v16bf condA = ldA(&h[ln][0], 0, hf);   // regs; survives h overwrite

    v8f t0 = {}, t1 = {};
#pragma unroll 1
    for (int m = 0; m < 2; ++m) {          // m=0: t-MLP, m=1: s-MLP
      const unsigned short* wb  = pw + (size_t)(i * 2 + m) * 294912u;
      const unsigned short* w1p = wb;
      const unsigned short* w2p = wb + 16384;
      const unsigned short* w3p = wb + 278528;
      const float* b1p = (m ? bs1 : bt1) + i * 512;
      const float* b2p = (m ? bs2 : bt2) + i * 512;
      const float* b3p = (m ? bs3 : bt3) + i * 32;

      // ---- GEMM1: h1 = relu(z1 @ W1 + b1)  [16x32]@[32x512] ----
#pragma unroll 1
      for (int nt = 0; nt < 32; ++nt) {
        float bv = b1p[nt * 16 + ln];
        v8f acc = {bv, bv, bv, bv, bv, bv, bv, bv};
        acc = wmma_bf16(condA, ldB(w1p + (size_t)nt * 512, lane), acc);
        int nc = nt * 16 + ln;
#pragma unroll
        for (int v = 0; v < 8; ++v)
          h[v + 8 * hf][nc] = (unsigned short)(relu_bf16_bits(acc[v]) >> 16);
      }

      // ---- GEMM2: h2 = relu(h1 @ W2 + b2), written in place over h1 ----
      v16bf aR[16];                        // all of h1 register-cached
#pragma unroll
      for (int kt = 0; kt < 16; ++kt) aR[kt] = ldA(&h[ln][0], kt, hf);
#pragma unroll 1
      for (int nt = 0; nt < 32; ++nt) {
        float bv = b2p[nt * 16 + ln];
        v8f acc = {bv, bv, bv, bv, bv, bv, bv, bv};
        const unsigned short* wn = w2p + (size_t)nt * 16 * 512;
#pragma unroll
        for (int kt = 0; kt < 16; ++kt)
          acc = wmma_bf16(aR[kt], ldB(wn + (size_t)kt * 512, lane), acc);
        int nc = nt * 16 + ln;
#pragma unroll
        for (int v = 0; v < 8; ++v)
          h[v + 8 * hf][nc] = (unsigned short)(relu_bf16_bits(acc[v]) >> 16);
      }

      // ---- GEMM3: out = h2 @ W3 + b3  [16x512]@[512x32] ----
#pragma unroll
      for (int kt = 0; kt < 16; ++kt) aR[kt] = ldA(&h[ln][0], kt, hf);
      v8f res[2];
#pragma unroll
      for (int nt = 0; nt < 2; ++nt) {
        float bv = b3p[nt * 16 + ln];
        v8f acc = {bv, bv, bv, bv, bv, bv, bv, bv};
        const unsigned short* wn = w3p + (size_t)nt * 16 * 512;
#pragma unroll
        for (int kt = 0; kt < 16; ++kt)
          acc = wmma_bf16(aR[kt], ldB(wn + (size_t)kt * 512, lane), acc);
        res[nt] = acc;
      }

      if (m == 0) {
        t0 = res[0]; t1 = res[1];
      } else {
        // ---- s = tanh(.)*scale+shift ; z0 = z0*exp(s)+t ; logdet += sum(s)
        float ssum[8];
#pragma unroll
        for (int v = 0; v < 8; ++v) ssum[v] = 0.0f;
#pragma unroll
        for (int nt = 0; nt < 2; ++nt) {
          int n = nt * 16 + ln;                 // C layout: N = ln + 16*nt
          float sc = s_scale[i * 32 + n];
          float sh = s_shift[i * 32 + n];
#pragma unroll
          for (int v = 0; v < 8; ++v) {         // C layout: M = v + 8*hf
            float sv = tanhf(res[nt][v]) * sc + sh;
            int r  = v + 8 * hf;
            int cz = 2 * n + off0;
            float tv = (nt == 0) ? t0[v] : t1[v];
            zbuf[r][cz] = zbuf[r][cz] * __expf(sv) + tv;
            ssum[v] += sv;
          }
        }
#pragma unroll
        for (int v = 0; v < 8; ++v) {           // reduce over 16 lanes (N dim)
          float x = ssum[v];
          x += __shfl_xor(x, 1, 32);
          x += __shfl_xor(x, 2, 32);
          x += __shfl_xor(x, 4, 32);
          x += __shfl_xor(x, 8, 32);
          logacc[v] += x;
        }
      }
    } // mlp
  } // layers

  // ---- write z tile and log_det ----
  {
    int r  = lane >> 1;
    int cb = (lane & 1) * 32;
    float4* dst = (float4*)(outZ + (size_t)(row0 + r) * 64 + cb);
    const float4* src = (const float4*)&zbuf[r][cb];
#pragma unroll
    for (int j = 0; j < 8; ++j) dst[j] = src[j];
  }
  if (ln == 0) {
#pragma unroll
    for (int v = 0; v < 8; ++v)
      outLD[row0 + v + 8 * hf] = logacc[v];
  }
}

// ---------------------------------------------------------------------------
extern "C" void kernel_launch(void* const* d_in, const int* in_sizes, int n_in,
                              void* d_out, int out_size, void* d_ws, size_t ws_size,
                              hipStream_t stream) {
  (void)in_sizes; (void)n_in; (void)out_size; (void)ws_size;
  const float* y   = (const float*)d_in[0];
  const float* Wt1 = (const float*)d_in[1];
  const float* bt1 = (const float*)d_in[2];
  const float* Wt2 = (const float*)d_in[3];
  const float* bt2 = (const float*)d_in[4];
  const float* Wt3 = (const float*)d_in[5];
  const float* bt3 = (const float*)d_in[6];
  const float* Ws1 = (const float*)d_in[7];
  const float* bs1 = (const float*)d_in[8];
  const float* Ws2 = (const float*)d_in[9];
  const float* bs2 = (const float*)d_in[10];
  const float* Ws3 = (const float*)d_in[11];
  const float* bs3 = (const float*)d_in[12];
  const float* s_scale = (const float*)d_in[13];
  const float* s_shift = (const float*)d_in[14];

  unsigned short* pw = (unsigned short*)d_ws;
  const int LSTR = 589824;            // bf16 elems per layer (both MLPs)
  const int OFF_S = 294912;           // s-MLP offset within a layer
  const int OFF_W2 = 16384, OFF_W3 = 278528;

  pack_w<<< (8 * 1  * 32 * 32) / 256, 256, 0, stream>>>(Wt1, pw,                   32, 512, LSTR);
  pack_w<<< (8 * 16 * 32 * 32) / 256, 256, 0, stream>>>(Wt2, pw + OFF_W2,         512, 512, LSTR);
  pack_w<<< (8 * 16 * 2  * 32) / 256, 256, 0, stream>>>(Wt3, pw + OFF_W3,         512,  32, LSTR);
  pack_w<<< (8 * 1  * 32 * 32) / 256, 256, 0, stream>>>(Ws1, pw + OFF_S,           32, 512, LSTR);
  pack_w<<< (8 * 16 * 32 * 32) / 256, 256, 0, stream>>>(Ws2, pw + OFF_S + OFF_W2, 512, 512, LSTR);
  pack_w<<< (8 * 16 * 2  * 32) / 256, 256, 0, stream>>>(Ws3, pw + OFF_S + OFF_W3, 512,  32, LSTR);

  float* outZ  = (float*)d_out;
  float* outLD = outZ + (size_t)65536 * 64;
  realnvp_fused<<<65536 / ROWS, 32, 0, stream>>>(y, bt1, bt2, bt3, bs1, bs2, bs3,
                                                 s_scale, s_shift, pw, outZ, outLD);
}
